// TABGNN_82643760709805
// MI455X (gfx1250) — matrix operand order repaired and checked by reading, not compiled, not run on hardware
//
#include <hip/hip_runtime.h>
#include <math.h>

#define CH 128
#define FFDIM 2048
#define NLAYER 2
#define N_NODES 8000
#define N_EDGES 20000
#define B_TGT 2048
#define ROWS (B_TGT + N_EDGES)   /* 22048 */
#define TOKENS (ROWS * 4)        /* 88192 */
#define EPSLN 1e-5f

typedef __attribute__((ext_vector_type(16))) _Float16 v16h;
typedef __attribute__((ext_vector_type(8)))  float    v8f;

static __device__ inline v8f wmma16(v16h a, v16h b, v8f c) {
  return __builtin_amdgcn_wmma_f32_16x16x32_f16(false, a, false, b, (short)0, c, false, false);
}
static __device__ inline v8f zero8() {
  v8f z = {0.f,0.f,0.f,0.f,0.f,0.f,0.f,0.f};
  return z;
}

// ---- WMMA fragment loaders (wave32 layouts per CDNA5 ISA 7.12.2) ----
// A 16x32 f16: lane l (m=l&15, half h=l>>4): VGPR v holds K = (v>>2)*16 + h*8 + (v&3)*2 (+1)
static __device__ inline v16h load_a_f16g(const _Float16* __restrict__ A, int lda, int m0, int k0) {
  int l = threadIdx.x & 31;
  const _Float16* row = A + (size_t)(m0 + (l & 15)) * (size_t)lda;
  int kb = k0 + ((l >> 4) << 3);
  v16h a;
#pragma unroll
  for (int v = 0; v < 8; ++v) {
    int k = kb + ((v >> 2) << 4) + ((v & 3) << 1);
    a[2*v]   = row[k];
    a[2*v+1] = row[k+1];
  }
  return a;
}
static __device__ inline v16h load_a_f16lds(const _Float16* __restrict__ P, int ldk, int k0) {
  int l = threadIdx.x & 31;
  const _Float16* row = P + (size_t)(l & 15) * (size_t)ldk;
  int kb = k0 + ((l >> 4) << 3);
  v16h a;
#pragma unroll
  for (int v = 0; v < 8; ++v) {
    int k = kb + ((v >> 2) << 4) + ((v & 3) << 1);
    a[2*v]   = row[k];
    a[2*v+1] = row[k+1];
  }
  return a;
}
// B 32x16 f16 from weight W (N,K) row-major: B[k][n] = W[n][k].
// lane l (n=l&15, half h=l>>4): VGPR v holds K = (v>>2)*16 + h*8 + (v&3)*2 (+1)
static __device__ inline v16h load_b_f16(const _Float16* __restrict__ W, int ldk, int n0, int k0) {
  int l = threadIdx.x & 31;
  const _Float16* row = W + (size_t)(n0 + (l & 15)) * (size_t)ldk;
  int kb = k0 + ((l >> 4) << 3);
  v16h b;
#pragma unroll
  for (int v = 0; v < 8; ++v) {
    int k = kb + ((v >> 2) << 4) + ((v & 3) << 1);
    b[2*v]   = row[k];
    b[2*v+1] = row[k+1];
  }
  return b;
}
// C/D 16x16 f32: lane l (n=l&15, h=l>>4), VGPR r -> m = 8h + r
static __device__ inline void store_c(void* __restrict__ Cp, int ldc, int m0, int n0,
                                      v8f c, const float* __restrict__ bias, int act, int out_half) {
  int l = threadIdx.x & 31;
  int n = n0 + (l & 15);
  int mb = m0 + ((l >> 4) << 3);
  float bv = bias ? bias[n] : 0.f;
#pragma unroll
  for (int r = 0; r < 8; ++r) {
    float v = c[r] + bv;
    if (act == 1) v = v > 0.f ? v : 0.f;
    size_t idx = (size_t)(mb + r) * (size_t)ldc + n;
    if (out_half) ((_Float16*)Cp)[idx] = (_Float16)v;
    else          ((float*)Cp)[idx] = v;
  }
}

// ---- weight pre-conversion ----
__global__ void k_f32tof16(const float* __restrict__ in, _Float16* __restrict__ out, size_t n) {
  size_t i = (size_t)blockIdx.x * blockDim.x + threadIdx.x;
  if (i < n) out[i] = (_Float16)in[i];
}

// ---- generic GEMM: C(M,N) = act(A(M,K)f16 @ W(N,K)f16^T + bias). 32x32 tile / wave ----
__global__ void k_gemm(const _Float16* __restrict__ A, const _Float16* __restrict__ W,
                       const float* __restrict__ bias, void* __restrict__ C,
                       int M, int N, int K, int act, int out_half) {
  int wave = (int)((blockIdx.x * blockDim.x + threadIdx.x) >> 5);
  int tn = N >> 5;
  if (wave >= (M >> 5) * tn) return;
  int m0 = (wave / tn) << 5, n0 = (wave % tn) << 5;
  v8f c00 = zero8(), c01 = zero8(), c10 = zero8(), c11 = zero8();
  for (int k0 = 0; k0 < K; k0 += 32) {
    v16h a0 = load_a_f16g(A, K, m0, k0);
    v16h a1 = load_a_f16g(A, K, m0 + 16, k0);
    v16h b0 = load_b_f16(W, K, n0, k0);
    v16h b1 = load_b_f16(W, K, n0 + 16, k0);
    c00 = wmma16(a0, b0, c00);
    c01 = wmma16(a0, b1, c01);
    c10 = wmma16(a1, b0, c10);
    c11 = wmma16(a1, b1, c11);
  }
  store_c(C, N, m0,      n0,      c00, bias, act, out_half);
  store_c(C, N, m0,      n0 + 16, c01, bias, act, out_half);
  store_c(C, N, m0 + 16, n0,      c10, bias, act, out_half);
  store_c(C, N, m0 + 16, n0 + 16, c11, bias, act, out_half);
}

// ---- fused FF: OUT = relu(H @ W1^T + b1) @ W2^T + b2, FF dim streamed through LDS ----
__global__ __launch_bounds__(128) void k_ff_fused(const _Float16* __restrict__ H,
    const _Float16* __restrict__ W1, const float* __restrict__ B1,
    const _Float16* __restrict__ W2, const float* __restrict__ B2,
    float* __restrict__ OUT, int M) {
  __shared__ _Float16 sP[4][16 * CH];
  int w = threadIdx.x >> 5;
  int m0 = (blockIdx.x * 4 + w) * 16;
  if (m0 >= M) return;
  int l = threadIdx.x & 31;
  _Float16* P = sP[w];
  v16h a[4];
#pragma unroll
  for (int kc = 0; kc < 4; ++kc) a[kc] = load_a_f16g(H, CH, m0, kc * 32);
  v8f acc[8];
#pragma unroll
  for (int nt = 0; nt < 8; ++nt) acc[nt] = zero8();
  for (int fc = 0; fc < FFDIM / CH; ++fc) {
    // stage 1: P = relu(H @ W1_chunk^T + b1_chunk), 16x128 -> LDS f16
#pragma unroll
    for (int nt = 0; nt < 8; ++nt) {
      v8f p = zero8();
#pragma unroll
      for (int kc = 0; kc < 4; ++kc) {
        v16h b = load_b_f16(W1, CH, fc * CH + nt * 16, kc * 32);
        p = wmma16(a[kc], b, p);
      }
      int n = nt * 16 + (l & 15);
      int mb = (l >> 4) << 3;
      float bv = B1[fc * CH + n];
#pragma unroll
      for (int r = 0; r < 8; ++r) {
        float v = p[r] + bv;
        v = v > 0.f ? v : 0.f;
        P[(mb + r) * CH + n] = (_Float16)v;
      }
    }
    __asm volatile("" ::: "memory");  // LDS is in-order per wave; block compiler reordering
    v16h pa[4];
#pragma unroll
    for (int kc = 0; kc < 4; ++kc) pa[kc] = load_a_f16lds(P, CH, kc * 32);
    __asm volatile("" ::: "memory");
    // stage 2: acc += P @ W2_chunk^T
#pragma unroll
    for (int nt = 0; nt < 8; ++nt) {
#pragma unroll
      for (int kc = 0; kc < 4; ++kc) {
        v16h b = load_b_f16(W2, FFDIM, nt * 16, fc * CH + kc * 32);
        acc[nt] = wmma16(pa[kc], b, acc[nt]);
      }
    }
  }
#pragma unroll
  for (int nt = 0; nt < 8; ++nt) store_c(OUT, CH, m0, nt * 16, acc[nt], B2, 0, 0);
}

// ---- small kernels ----
__global__ void k_node_embed(const float* __restrict__ x, const float* __restrict__ w,
                             const float* __restrict__ b, float* __restrict__ X, int n) {
  int i = blockIdx.x * blockDim.x + threadIdx.x;
  if (i >= n * CH) return;
  X[i] = x[i >> 7] * w[i & 127] + b[i & 127];
}

__global__ void k_build_tokens(const float* __restrict__ cls, const float* __restrict__ tgt,
                               const float* __restrict__ ea, float* __restrict__ T,
                               _Float16* __restrict__ T16, float* __restrict__ ORIG) {
  size_t i = (size_t)blockIdx.x * blockDim.x + threadIdx.x;
  if (i >= (size_t)ROWS * 512) return;
  int c = (int)(i & 127);
  int s = (int)((i >> 7) & 3);
  int r = (int)(i >> 9);
  float v;
  if (s == 0) v = cls[c];
  else if (r < B_TGT) v = tgt[((size_t)r * 3 + (s - 1)) * CH + c];
  else v = ea[((size_t)(r - B_TGT) * 3 + (s - 1)) * CH + c];
  T[i] = v; T16[i] = (_Float16)v; ORIG[i] = v;
}

__global__ void k_attn(const float* __restrict__ QKV, _Float16* __restrict__ O, int rows) {
  int wid = (int)((blockIdx.x * blockDim.x + threadIdx.x) >> 5);
  if (wid >= rows) return;
  int l = threadIdx.x & 31;
  int h = l >> 2, qi = l & 3;
  const float* base = QKV + (size_t)wid * 4 * 384;
  float q[16];
#pragma unroll
  for (int d = 0; d < 16; ++d) q[d] = base[qi * 384 + h * 16 + d];
  float s[4];
#pragma unroll
  for (int j = 0; j < 4; ++j) {
    float acc = 0.f;
#pragma unroll
    for (int d = 0; d < 16; ++d) acc += q[d] * base[j * 384 + 128 + h * 16 + d];
    s[j] = acc * 0.25f;  // 1/sqrt(16)
  }
  float mx = fmaxf(fmaxf(s[0], s[1]), fmaxf(s[2], s[3]));
  float e[4], se = 0.f;
#pragma unroll
  for (int j = 0; j < 4; ++j) { e[j] = expf(s[j] - mx); se += e[j]; }
  float inv = 1.f / se;
  _Float16* o = O + ((size_t)wid * 4 + qi) * CH + h * 16;
#pragma unroll
  for (int d = 0; d < 16; ++d) {
    float acc = 0.f;
#pragma unroll
    for (int j = 0; j < 4; ++j) acc += e[j] * base[j * 384 + 256 + h * 16 + d];
    o[d] = (_Float16)(acc * inv);
  }
}

static __device__ inline float wredsum(float v) {
#pragma unroll
  for (int o = 16; o > 0; o >>= 1) v += __shfl_xor(v, o);
  return v;
}

__global__ void k_add_ln(const float* __restrict__ A, const float* __restrict__ B,
                         const float* __restrict__ g, const float* __restrict__ bb,
                         float* __restrict__ Out, _Float16* __restrict__ Out16, int rows) {
  int wid = (int)((blockIdx.x * blockDim.x + threadIdx.x) >> 5);
  if (wid >= rows) return;
  int l = threadIdx.x & 31;
  const float* a = A + (size_t)wid * CH;
  const float* b2 = B + (size_t)wid * CH;
  float v[4];
#pragma unroll
  for (int j = 0; j < 4; ++j) v[j] = a[j * 32 + l] + b2[j * 32 + l];
  float mu = wredsum(v[0] + v[1] + v[2] + v[3]) * (1.f / 128.f);
  float q = 0.f;
#pragma unroll
  for (int j = 0; j < 4; ++j) { float d = v[j] - mu; q += d * d; }
  float rs = rsqrtf(wredsum(q) * (1.f / 128.f) + EPSLN);
#pragma unroll
  for (int j = 0; j < 4; ++j) {
    float o = (v[j] - mu) * rs * g[j * 32 + l] + bb[j * 32 + l];
    Out[(size_t)wid * CH + j * 32 + l] = o;
    Out16[(size_t)wid * CH + j * 32 + l] = (_Float16)o;
  }
}

__global__ void k_ff_combine(const float* __restrict__ H, const float* __restrict__ FFv,
                             float* __restrict__ T, _Float16* __restrict__ T16,
                             const float* __restrict__ g2, const float* __restrict__ b2,
                             const float* __restrict__ gT, const float* __restrict__ bT,
                             int rows) {
  int wid = (int)((blockIdx.x * blockDim.x + threadIdx.x) >> 5);
  if (wid >= rows) return;
  int l = threadIdx.x & 31;
  const float* h = H + (size_t)wid * CH;
  const float* f = FFv + (size_t)wid * CH;
  float* t = T + (size_t)wid * CH;
  _Float16* t16 = T16 + (size_t)wid * CH;
  float v[4];
#pragma unroll
  for (int j = 0; j < 4; ++j) v[j] = h[j * 32 + l] + f[j * 32 + l];
  float mu = wredsum(v[0] + v[1] + v[2] + v[3]) * (1.f / 128.f);
  float q = 0.f;
#pragma unroll
  for (int j = 0; j < 4; ++j) { float d = v[j] - mu; q += d * d; }
  float rs = rsqrtf(wredsum(q) * (1.f / 128.f) + EPSLN);
#pragma unroll
  for (int j = 0; j < 4; ++j) v[j] = (v[j] - mu) * rs * g2[j * 32 + l] + b2[j * 32 + l];
  mu = wredsum(v[0] + v[1] + v[2] + v[3]) * (1.f / 128.f);
  q = 0.f;
#pragma unroll
  for (int j = 0; j < 4; ++j) { float d = v[j] - mu; q += d * d; }
  rs = rsqrtf(wredsum(q) * (1.f / 128.f) + EPSLN);
#pragma unroll
  for (int j = 0; j < 4; ++j) {
    float h3 = (v[j] - mu) * rs * gT[j * 32 + l] + bT[j * 32 + l];
    float nv = (t[j * 32 + l] + h3) * 0.5f;
    t[j * 32 + l] = nv;
    t16[j * 32 + l] = (_Float16)nv;
  }
}

__global__ void k_combine_half(const float* __restrict__ ORIG, float* __restrict__ T,
                               _Float16* __restrict__ T16, size_t n) {
  size_t i = (size_t)blockIdx.x * blockDim.x + threadIdx.x;
  if (i < n) {
    float v = (ORIG[i] + T[i]) * 0.5f;
    T[i] = v;
    T16[i] = (_Float16)v;
  }
}

__global__ void k_split_emb(const float* __restrict__ EMB, float* __restrict__ out,
                            float* __restrict__ EA, _Float16* __restrict__ EA16) {
  size_t i = (size_t)blockIdx.x * blockDim.x + threadIdx.x;
  if (i >= (size_t)ROWS * CH) return;
  int r = (int)(i >> 7);
  float v = EMB[i];
  if (r < B_TGT) out[(size_t)(N_NODES + N_EDGES) * CH + i] = v;   // tgt section
  else {
    size_t j = i - (size_t)B_TGT * CH;
    EA[j] = v;
    EA16[j] = (_Float16)v;
  }
}

__global__ void k_gather3(const float* __restrict__ X, const float* __restrict__ E,
                          const int* __restrict__ idxA, const int* __restrict__ idxB,
                          _Float16* __restrict__ OUT, int nE) {
  size_t i = (size_t)blockIdx.x * blockDim.x + threadIdx.x;
  if (i >= (size_t)nE * 384) return;
  int c = (int)(i % 384);
  int e = (int)(i / 384);
  float v;
  if (c < 128)      v = X[(size_t)idxA[e] * CH + c];
  else if (c < 256) v = X[(size_t)idxB[e] * CH + (c - 128)];
  else              v = E[(size_t)e * CH + (c - 256)];
  OUT[i] = (_Float16)v;
}

static __device__ inline unsigned fenc(float f) {
  unsigned u = __float_as_uint(f);
  return (u & 0x80000000u) ? ~u : (u | 0x80000000u);
}
static __device__ inline float fdec(unsigned k) {
  unsigned u = (k & 0x80000000u) ? (k & 0x7FFFFFFFu) : ~k;
  return __uint_as_float(u);
}

__global__ void k_init(float* SUM, float* SSQ, unsigned* MAXE, unsigned* MINE,
                       float* CNT, float* SCAL, float* BNS, float* BNS2) {
  int i = blockIdx.x * blockDim.x + threadIdx.x;
  if (i < N_NODES * CH) { SUM[i] = 0.f; SSQ[i] = 0.f; MAXE[i] = 0u; MINE[i] = 0xFFFFFFFFu; }
  if (i < N_NODES) CNT[i] = 0.f;
  if (i < CH) { BNS[i] = 0.f; BNS2[i] = 0.f; }
  if (i == 0) SCAL[0] = 0.f;
}

__global__ void k_scatter(const float* __restrict__ Mm, const int* __restrict__ dst,
                          float* SUM, float* SSQ, unsigned* MAXE, unsigned* MINE,
                          float* CNT, int nE) {
  size_t i = (size_t)blockIdx.x * blockDim.x + threadIdx.x;
  if (i >= (size_t)nE * CH) return;
  int c = (int)(i & 127);
  int e = (int)(i >> 7);
  int d = dst[e];
  float v = Mm[i];
  size_t o = (size_t)d * CH + c;
  atomicAdd(&SUM[o], v);
  atomicAdd(&SSQ[o], v * v);
  atomicMax(&MAXE[o], fenc(v));
  atomicMin(&MINE[o], fenc(v));
  if (c == 0) atomicAdd(&CNT[d], 1.0f);
}

__global__ void k_avglog(const float* __restrict__ CNT, float* SCAL, int n) {
  int i = blockIdx.x * blockDim.x + threadIdx.x;
  if (i < n) atomicAdd(&SCAL[0], logf(CNT[i] + 1.0f));
}

__global__ void k_aggf(const float* __restrict__ X, const float* __restrict__ SUM,
                       const float* __restrict__ SSQ, const unsigned* __restrict__ MAXE,
                       const unsigned* __restrict__ MINE, const float* __restrict__ CNT,
                       const float* __restrict__ SCAL, _Float16* __restrict__ AGGF, int nN) {
  size_t i = (size_t)blockIdx.x * blockDim.x + threadIdx.x;
  if (i >= (size_t)nN * 1664) return;
  int c = (int)(i % 1664);
  int n = (int)(i / 1664);
  float v;
  if (c < CH) {
    v = X[(size_t)n * CH + c];
  } else {
    int cc = c - CH;
    int grp = cc / 512;
    int cb = cc % 512;
    int stat = cb >> 7;
    int ch = cb & 127;
    float cnt = CNT[n];
    float deg = fmaxf(cnt, 1.0f);
    size_t o = (size_t)n * CH + ch;
    float base;
    if (stat == 0) base = SUM[o] / deg;
    else if (stat == 1) base = (cnt > 0.f) ? fdec(MAXE[o]) : 0.f;
    else if (stat == 2) base = (cnt > 0.f) ? fdec(MINE[o]) : 0.f;
    else {
      float mean = SUM[o] / deg;
      float msq = SSQ[o] / deg;
      float d = msq - mean * mean;
      base = sqrtf((d > 0.f ? d : 0.f) + 1e-5f);
    }
    float ld = logf(deg + 1.f);
    float al = SCAL[0] * (1.f / (float)N_NODES);
    float fct = (grp == 0) ? 1.f : ((grp == 1) ? (ld / al) : (al / ld));
    v = base * fct;
  }
  AGGF[i] = (_Float16)v;
}

__global__ void k_bnstats(const float* __restrict__ O2, float* BNS, float* BNS2, int nN) {
  size_t i = (size_t)blockIdx.x * blockDim.x + threadIdx.x;
  if (i >= (size_t)nN * CH) return;
  float v = O2[i];
  int c = (int)(i & 127);
  atomicAdd(&BNS[c], v);
  atomicAdd(&BNS2[c], v * v);
}

__global__ void k_bnapply(const float* __restrict__ O2, const float* __restrict__ BNS,
                          const float* __restrict__ BNS2, const float* __restrict__ g,
                          const float* __restrict__ b, float* __restrict__ X, int nN) {
  size_t i = (size_t)blockIdx.x * blockDim.x + threadIdx.x;
  if (i >= (size_t)nN * CH) return;
  int c = (int)(i & 127);
  float mu = BNS[c] * (1.f / (float)N_NODES);
  float var = BNS2[c] * (1.f / (float)N_NODES) - mu * mu;
  float o = (O2[i] - mu) * rsqrtf(var + EPSLN) * g[c] + b[c];
  X[i] = (X[i] + fmaxf(o, 0.f)) * 0.5f;
}

__global__ void k_ea_update(float* __restrict__ EA, _Float16* __restrict__ EA16,
                            const float* __restrict__ EU, size_t n) {
  size_t i = (size_t)blockIdx.x * blockDim.x + threadIdx.x;
  if (i < n) {
    float v = EA[i] + EU[i] * 0.5f;
    EA[i] = v;
    EA16[i] = (_Float16)v;
  }
}

__global__ void k_final_copy(const float* __restrict__ X, const float* __restrict__ EA,
                             float* __restrict__ out) {
  size_t i = (size_t)blockIdx.x * blockDim.x + threadIdx.x;
  size_t nx = (size_t)N_NODES * CH;
  if (i < nx) out[i] = X[i];
  else if (i < nx + (size_t)N_EDGES * CH) out[i] = EA[i - nx];
}

extern "C" void kernel_launch(void* const* d_in, const int* in_sizes, int n_in,
                              void* d_out, int out_size, void* d_ws, size_t ws_size,
                              hipStream_t stream) {
  (void)in_sizes; (void)n_in; (void)out_size; (void)ws_size;
  auto F = [&](int i) { return (const float*)d_in[i]; };
  const float* xin  = F(0);
  const int* eidx   = (const int*)d_in[1];
  const int* src    = eidx;
  const int* dst    = eidx + N_EDGES;
  const float* eattr = F(2);
  const float* tattr = F(3);
  const float* cls   = F(4);

  auto blk = [](size_t n) { return dim3((unsigned)((n + 255) / 256)); };

  // ---- workspace layout (byte-based, 256B aligned) ----
  char* base = (char*)d_ws;
  size_t off = 0;
  auto allocb = [&](size_t bytes) { void* p = base + off; off += (bytes + 255) & ~(size_t)255; return p; };
  float*    ORIG = (float*)   allocb((size_t)ROWS * 512 * 4);
  float*    T32  = (float*)   allocb((size_t)ROWS * 512 * 4);
  _Float16* T16  = (_Float16*)allocb((size_t)ROWS * 512 * 2);
  float*    EMB  = (float*)   allocb((size_t)ROWS * CH * 4);
  float*    X    = (float*)   allocb((size_t)N_NODES * CH * 4);
  float*    EA   = (float*)   allocb((size_t)N_EDGES * CH * 4);
  _Float16* EA16 = (_Float16*)allocb((size_t)N_EDGES * CH * 2);
  _Float16* WH   = (_Float16*)allocb((size_t)2100000 * 2);   // f16 weight pool
  char* SCR = base + off;

  // tab-phase scratch
  {
    // laid out below via pointers into SCR
  }
  float*    QKV  = (float*)SCR;
  _Float16* O16  = (_Float16*)((char*)QKV  + (size_t)TOKENS * 384 * 4);
  float*    OUTP = (float*)   ((char*)O16  + (size_t)TOKENS * CH * 2);
  float*    H32  = (float*)   ((char*)OUTP + (size_t)TOKENS * CH * 4);
  _Float16* H16  = (_Float16*)((char*)H32  + (size_t)TOKENS * CH * 4);
  float*    FFB  = (float*)   ((char*)H16  + (size_t)TOKENS * CH * 2);
  // gnn-phase scratch (reuses SCR)
  float*    E     = (float*)SCR;
  _Float16* MIN16 = (_Float16*)((char*)E     + (size_t)N_EDGES * CH * 4);
  float*    Mm    = (float*)   ((char*)MIN16 + (size_t)N_EDGES * 384 * 2);
  float*    SUM   = (float*)   ((char*)Mm    + (size_t)N_EDGES * CH * 4);
  float*    SSQ   = SUM + (size_t)N_NODES * CH;
  unsigned* MAXE  = (unsigned*)(SSQ + (size_t)N_NODES * CH);
  unsigned* MINE  = MAXE + (size_t)N_NODES * CH;
  float*    CNT   = (float*)(MINE + (size_t)N_NODES * CH);
  float*    SCAL  = CNT + N_NODES;
  float*    BNS   = SCAL + 4;
  float*    BNS2  = BNS + CH;
  _Float16* AGGF  = (_Float16*)(BNS2 + CH);
  _Float16* O1_16 = AGGF + (size_t)N_NODES * 1664;
  float*    OUT2  = (float*)(O1_16 + (size_t)N_NODES * CH);
  _Float16* EUIN  = (_Float16*)(OUT2 + (size_t)N_NODES * CH);
  _Float16* EU1B  = EUIN + (size_t)N_EDGES * 384;
  float*    EU2B  = (float*)(EU1B + (size_t)N_EDGES * CH);

  // ---- pre-convert all GEMM weights to f16 (once per launch, stream-ordered) ----
  size_t woff = 0;
  auto cvtw = [&](int idx, size_t n) {
    _Float16* p = WH + woff; woff += n;
    k_f32tof16<<<blk(n), 256, 0, stream>>>(F(idx), p, n);
    return (const _Float16*)p;
  };
  const _Float16 *qkvW[NLAYER], *outW[NLAYER], *ff1W[NLAYER], *ff2W[NLAYER];
  const _Float16 *encW[NLAYER], *preW[NLAYER], *postW[NLAYER], *linW[NLAYER];
  const _Float16 *eu1W[NLAYER], *eu2W[NLAYER];
  const _Float16* embW = cvtw(7, (size_t)CH * 512);
  for (int li = 0; li < NLAYER; ++li) {
    int b = 9 + li * 14;
    qkvW[li] = cvtw(b + 0, (size_t)384 * CH);
    outW[li] = cvtw(b + 2, (size_t)CH * CH);
    ff1W[li] = cvtw(b + 4, (size_t)FFDIM * CH);
    ff2W[li] = cvtw(b + 6, (size_t)CH * FFDIM);
  }
  for (int li = 0; li < NLAYER; ++li) {
    int b = 9 + NLAYER * 14 + li * 14;
    encW[li]  = cvtw(b + 0,  (size_t)CH * CH);
    preW[li]  = cvtw(b + 2,  (size_t)CH * 384);
    postW[li] = cvtw(b + 4,  (size_t)CH * 1664);
    linW[li]  = cvtw(b + 6,  (size_t)CH * CH);
    eu1W[li]  = cvtw(b + 10, (size_t)CH * 384);
    eu2W[li]  = cvtw(b + 12, (size_t)CH * CH);
  }

  auto gemm = [&](const _Float16* A, const _Float16* W, const float* b, void* C,
                  int M, int N, int K, int act, int out_half) {
    int waves = (M >> 5) * (N >> 5);
    k_gemm<<<dim3((unsigned)((waves + 7) / 8)), dim3(256), 0, stream>>>(A, W, b, C, M, N, K,
                                                                        act, out_half);
  };

  // ---- embeddings + token build ----
  k_node_embed<<<blk((size_t)N_NODES * CH), 256, 0, stream>>>(xin, F(5), F(6), X, N_NODES);
  k_build_tokens<<<blk((size_t)ROWS * 512), 256, 0, stream>>>(cls, tattr, eattr, T32, T16, ORIG);

  // ---- tabular transformer layers ----
  for (int li = 0; li < NLAYER; ++li) {
    int b = 9 + li * 14;
    gemm(T16, qkvW[li], F(b + 1), QKV, TOKENS, 384, CH, 0, 0);            // qkv
    k_attn<<<dim3(ROWS / 8), 256, 0, stream>>>(QKV, O16, ROWS);
    gemm(O16, outW[li], F(b + 3), OUTP, TOKENS, CH, CH, 0, 0);            // out proj
    k_add_ln<<<dim3(TOKENS / 8), 256, 0, stream>>>(T32, OUTP, F(b + 8), F(b + 9), H32, H16, TOKENS);
    k_ff_fused<<<dim3(TOKENS / 64), 128, 0, stream>>>(H16, ff1W[li], F(b + 5),
                                                      ff2W[li], F(b + 7), FFB, TOKENS);
    k_ff_combine<<<dim3(TOKENS / 8), 256, 0, stream>>>(H32, FFB, T32, T16, F(b + 10), F(b + 11),
                                                       F(b + 12), F(b + 13), TOKENS);
  }
  k_combine_half<<<blk((size_t)ROWS * 512), 256, 0, stream>>>(ORIG, T32, T16, (size_t)ROWS * 512);
  gemm(T16, embW, F(8), EMB, ROWS, CH, 512, 0, 0);                        // edge_emb
  k_split_emb<<<blk((size_t)ROWS * CH), 256, 0, stream>>>(EMB, (float*)d_out, EA, EA16);

  // ---- PNA GNN layers ----
  for (int li = 0; li < NLAYER; ++li) {
    int b = 9 + NLAYER * 14 + li * 14;
    gemm(EA16, encW[li], F(b + 1), E, N_EDGES, CH, CH, 0, 0);             // edge_enc
    k_gather3<<<blk((size_t)N_EDGES * 384), 256, 0, stream>>>(X, E, dst, src, MIN16, N_EDGES);
    gemm(MIN16, preW[li], F(b + 3), Mm, N_EDGES, CH, 384, 0, 0);          // pre
    k_init<<<blk((size_t)N_NODES * CH), 256, 0, stream>>>(SUM, SSQ, MAXE, MINE, CNT, SCAL, BNS, BNS2);
    k_scatter<<<blk((size_t)N_EDGES * CH), 256, 0, stream>>>(Mm, dst, SUM, SSQ, MAXE, MINE, CNT, N_EDGES);
    k_avglog<<<blk((size_t)N_NODES), 256, 0, stream>>>(CNT, SCAL, N_NODES);
    k_aggf<<<blk((size_t)N_NODES * 1664), 256, 0, stream>>>(X, SUM, SSQ, MAXE, MINE, CNT, SCAL, AGGF, N_NODES);
    gemm(AGGF, postW[li], F(b + 5), O1_16, N_NODES, CH, 1664, 0, 1);      // post -> f16
    gemm(O1_16, linW[li], F(b + 7), OUT2, N_NODES, CH, CH, 0, 0);         // lin
    k_bnstats<<<blk((size_t)N_NODES * CH), 256, 0, stream>>>(OUT2, BNS, BNS2, N_NODES);
    k_bnapply<<<blk((size_t)N_NODES * CH), 256, 0, stream>>>(OUT2, BNS, BNS2, F(b + 8), F(b + 9), X, N_NODES);
    k_gather3<<<blk((size_t)N_EDGES * 384), 256, 0, stream>>>(X, EA, src, dst, EUIN, N_EDGES);
    gemm(EUIN, eu1W[li], F(b + 11), EU1B, N_EDGES, CH, 384, 1, 1);        // eu1 + relu -> f16
    gemm(EU1B, eu2W[li], F(b + 13), EU2B, N_EDGES, CH, CH, 0, 0);         // eu2
    k_ea_update<<<blk((size_t)N_EDGES * CH), 256, 0, stream>>>(EA, EA16, EU2B, (size_t)N_EDGES * CH);
  }
  k_final_copy<<<blk((size_t)(N_NODES + N_EDGES) * CH), 256, 0, stream>>>(X, EA, (float*)d_out);
}
